// PairUpdate_13280038879340
// MI455X (gfx1250) — matrix operand order repaired and checked by reading, not compiled, not run on hardware
//
#include <hip/hip_runtime.h>

typedef __attribute__((ext_vector_type(16))) __bf16 v16bf;
typedef __attribute__((ext_vector_type(8)))  __bf16 v8bf;
typedef __attribute__((ext_vector_type(4)))  __bf16 v4bf;
typedef __attribute__((ext_vector_type(8)))  float  v8f;
typedef __attribute__((ext_vector_type(4)))  float  v4f;

#define NSEQ   256
#define CZV    128
#define NNPOS  65536
#define INV_SQRT_CH 0.17677669529663689f

// ---------- wave32 reduction helpers ----------
__device__ __forceinline__ float wsum32(float v) {
#pragma unroll
  for (int m = 16; m >= 1; m >>= 1) v += __shfl_xor(v, m, 32);
  return v;
}
__device__ __forceinline__ float hsum16(float v) {
#pragma unroll
  for (int m = 8; m >= 1; m >>= 1) v += __shfl_xor(v, m, 32);
  return v;
}
__device__ __forceinline__ float hmax16(float v) {
#pragma unroll
  for (int m = 8; m >= 1; m >>= 1) v = fmaxf(v, __shfl_xor(v, m, 32));
  return v;
}

// ---------- WMMA operand loaders ----------
// A operand (16x32 bf16, row-major source, leading dim ld):
// lane L -> row row0+(L&15); K offsets {0..7,16..23} for L<16, {8..15,24..31} for L>=16.
__device__ __forceinline__ v16bf load_tile_a(const __bf16* base, int row0, int col0, int ld, int lane) {
  int r = row0 + (lane & 15);
  int c = col0 + ((lane >> 4) << 3);
  const __bf16* p = base + (size_t)r * ld + c;
  v8bf lo = *(const v8bf*)p;
  v8bf hi = *(const v8bf*)(p + 16);
  v16bf out;
#pragma unroll
  for (int t = 0; t < 8; ++t) { out[t] = lo[t]; out[t + 8] = hi[t]; }
  return out;
}
// B operand (32x16 bf16). Source stores columns as rows (transposed, leading dim ld):
// lane L -> column n0+(L&15), K offsets {0..7,16..23} for L<16, {8..15,24..31} for L>=16.
__device__ __forceinline__ v16bf load_tile_b(const __bf16* base, int n0, int col0, int ld, int lane) {
  int r = n0 + (lane & 15);
  int c = col0 + ((lane >> 4) << 3);
  const __bf16* p = base + (size_t)r * ld + c;
  v8bf lo = *(const v8bf*)p;
  v8bf hi = *(const v8bf*)(p + 16);
  v16bf out;
#pragma unroll
  for (int t = 0; t < 8; ++t) { out[t] = lo[t]; out[t + 8] = hi[t]; }
  return out;
}
__device__ __forceinline__ v8f wmma_bf16(v16bf a, v16bf b, v8f c) {
  return __builtin_amdgcn_wmma_f32_16x16x32_bf16(false, a, false, b, (short)0, c, false, false);
}

// ---------- weight transpose + bf16 convert : Wt[o][i] = W[i][o] ----------
__global__ void prep_w(const float* __restrict__ W, __bf16* __restrict__ Wt) {
  int idx = blockIdx.x * blockDim.x + threadIdx.x;   // 16384
  int o = idx >> 7, i = idx & 127;
  Wt[(size_t)o * 128 + i] = (__bf16)W[(size_t)i * 128 + o];
}

// ---------- edge bias: eb[h][q][k] = dist_bias + third ----------
__global__ void edge_bias_kernel(const float* __restrict__ coords,
                                 const float* __restrict__ edge_embed,
                                 const float* __restrict__ w1, const float* __restrict__ g1,
                                 const float* __restrict__ b1, const float* __restrict__ w2,
                                 const float* __restrict__ ln3g, const float* __restrict__ ln3b,
                                 const float* __restrict__ w3, float* __restrict__ eb) {
  int wid = (blockIdx.x * blockDim.x + threadIdx.x) >> 5;   // position q*256+k
  int lane = threadIdx.x & 31;
  int qi = wid >> 8, kj = wid & 255;
  float dx = coords[qi * 3 + 0] - coords[kj * 3 + 0];
  float dy = coords[qi * 3 + 1] - coords[kj * 3 + 1];
  float dz = coords[qi * 3 + 2] - coords[kj * 3 + 2];
  float dist = sqrtf(dx * dx + dy * dy + dz * dz + 1e-12f);
  const float inv_sigma = 39.0f / 4.75f;
  // h channel = lane (CH=32)
  float hc = 0.0f;
  for (int nb = 0; nb < 39; ++nb) {
    float mu = 0.325f + 0.125f * (float)nb;     // linspace(0.325, 5.075, 39)
    float t = (dist - mu) * inv_sigma;
    hc += __expf(-t * t) * w1[nb * 32 + lane];
  }
  float mean = wsum32(hc) * (1.0f / 32.0f);
  float d = hc - mean;
  float var = wsum32(d * d) * (1.0f / 32.0f);
  float hn = d * rsqrtf(var + 1e-5f) * g1[lane] + b1[lane];
  float db0 = wsum32(hn * w2[lane * 4 + 0]);
  float db1 = wsum32(hn * w2[lane * 4 + 1]);
  float db2 = wsum32(hn * w2[lane * 4 + 2]);
  float db3 = wsum32(hn * w2[lane * 4 + 3]);
  // third = LN(edge_embed) @ w3
  v4f x = *(const v4f*)(edge_embed + (size_t)wid * 128 + lane * 4);
  float s = x[0] + x[1] + x[2] + x[3];
  float mean3 = wsum32(s) * (1.0f / 128.0f);
  float vv = 0.0f;
#pragma unroll
  for (int t = 0; t < 4; ++t) { float dd = x[t] - mean3; vv += dd * dd; }
  float rstd3 = rsqrtf(wsum32(vv) * (1.0f / 128.0f) + 1e-5f);
  float t0 = 0, t1 = 0, t2 = 0, t3 = 0;
#pragma unroll
  for (int t = 0; t < 4; ++t) {
    int c = lane * 4 + t;
    float zn = (x[t] - mean3) * rstd3 * ln3g[c] + ln3b[c];
    t0 += zn * w3[c * 4 + 0]; t1 += zn * w3[c * 4 + 1];
    t2 += zn * w3[c * 4 + 2]; t3 += zn * w3[c * 4 + 3];
  }
  t0 = wsum32(t0); t1 = wsum32(t1); t2 = wsum32(t2); t3 = wsum32(t3);
  if (lane == 0) {
    eb[0 * NNPOS + wid] = db0 + t0;
    eb[1 * NNPOS + wid] = db1 + t1;
    eb[2 * NNPOS + wid] = db2 + t2;
    eb[3 * NNPOS + wid] = db3 + t3;
  }
}

// ---------- LayerNorm rows -> bf16 (optionally reading transposed source) ----------
__global__ void ln_rows(const float* __restrict__ src, const float* __restrict__ g,
                        const float* __restrict__ b, __bf16* __restrict__ dst, int transpose) {
  int wid = (blockIdx.x * blockDim.x + threadIdx.x) >> 5;   // dest row
  int lane = threadIdx.x & 31;
  int sr = transpose ? (((wid & 255) << 8) | (wid >> 8)) : wid;
  v4f x = *(const v4f*)(src + (size_t)sr * 128 + lane * 4);
  float s = x[0] + x[1] + x[2] + x[3];
  float mean = wsum32(s) * (1.0f / 128.0f);
  float vv = 0.0f;
#pragma unroll
  for (int t = 0; t < 4; ++t) { float d = x[t] - mean; vv += d * d; }
  float rstd = rsqrtf(wsum32(vv) * (1.0f / 128.0f) + 1e-5f);
  v4bf o;
#pragma unroll
  for (int t = 0; t < 4; ++t) {
    int c = lane * 4 + t;
    o[t] = (__bf16)((x[t] - mean) * rstd * g[c] + b[c]);
  }
  *(v4bf*)(dst + (size_t)wid * 128 + lane * 4) = o;
}

// ---------- 65536x128 @ 128x128 projection GEMM, 64 rows (4 M-tiles) per wave ----------
// mode 0: bf16 out (Q,K)     mode 1: bf16 transposed out vt[i][c'][pos] (V)
// mode 2: sigmoid->bf16 (G)  mode 3: f32 out = res + D (residual, optional transposed store)
__global__ void __launch_bounds__(256) gemm_proj(const __bf16* __restrict__ A,
                                                 const __bf16* __restrict__ Bt, int mode,
                                                 __bf16* __restrict__ outb, __bf16* __restrict__ vt,
                                                 const float* __restrict__ res,
                                                 float* __restrict__ outf, int transpose) {
  int wid = (blockIdx.x * blockDim.x + threadIdx.x) >> 5;
  int lane = threadIdx.x & 31;
  int mgrp = wid >> 3, ntile = wid & 7;
  int r0 = mgrp << 6;          // 64 rows per wave
  int n0 = ntile << 4;
  v8f accs[4] = {};
#pragma unroll
  for (int kk = 0; kk < 4; ++kk) {
    v16bf b = load_tile_b(Bt, n0, kk * 32, 128, lane);   // reused by 4 M-tiles
#pragma unroll
    for (int mi = 0; mi < 4; ++mi) {
      v16bf a = load_tile_a(A, r0 + (mi << 4), kk * 32, 128, lane);
      accs[mi] = wmma_bf16(a, b, accs[mi]);
    }
  }
  int hi = lane >> 4;
  int n = lane & 15;
  int gn = n0 + n;
#pragma unroll
  for (int mi = 0; mi < 4; ++mi) {
    int rm = r0 + (mi << 4);
    if (mode == 1) {           // transposed packed store: 8 consecutive pos per lane
      v8bf pk;
#pragma unroll
      for (int j = 0; j < 8; ++j) pk[j] = (__bf16)accs[mi][j];
      int pos0 = (rm & 255) + (hi << 3);
      int iblk = rm >> 8;
      *(v8bf*)(vt + ((size_t)iblk * 128 + gn) * 256 + pos0) = pk;
    } else {
#pragma unroll
      for (int j = 0; j < 8; ++j) {
        int gm = rm + j + (hi << 3);
        float v = accs[mi][j];
        if (mode == 0) {
          outb[(size_t)gm * 128 + gn] = (__bf16)v;
        } else if (mode == 2) {
          float e = __expf(-v);
          outb[(size_t)gm * 128 + gn] = (__bf16)__builtin_amdgcn_rcpf(1.0f + e);
        } else {
          size_t dst = transpose ? ((size_t)(gm & 255) * (NSEQ * CZV) + (size_t)(gm >> 8) * CZV + gn)
                                 : ((size_t)gm * CZV + gn);
          outf[dst] = res[dst] + v;
        }
      }
    }
  }
}

// ---------- flash-style triangle attention, one wave per (i, head, 16-q tile) ----------
__global__ void __launch_bounds__(256) attn_kernel(const __bf16* __restrict__ Q,
                                                   const __bf16* __restrict__ K,
                                                   const __bf16* __restrict__ Vt,
                                                   const __bf16* __restrict__ G,
                                                   const float* __restrict__ eb,
                                                   const float* __restrict__ mask,
                                                   __bf16* __restrict__ OG, int transpose) {
  __shared__ __bf16 pbuf[8][16][32];
  int wid = (blockIdx.x * blockDim.x + threadIdx.x) >> 5;
  int lane = threadIdx.x & 31;
  int slot = threadIdx.x >> 5;
  int i = wid >> 6;
  int h = (wid >> 4) & 3;
  int q0 = (wid & 15) << 4;
  int rowbase = i << 8;
  int hi = lane >> 4;
  int n = lane & 15;

  v16bf aq = load_tile_a(Q, rowbase + q0, h * 32, 128, lane);
  const __bf16* vbase = Vt + (size_t)i * 128 * 256;

  v8f O0 = {}, O1 = {};
  float mrun[8], lrun[8];
#pragma unroll
  for (int j = 0; j < 8; ++j) { mrun[j] = -1e30f; lrun[j] = 0.0f; }

#pragma unroll 2
  for (int kb = 0; kb < 8; ++kb) {
    int k0 = kb << 5;
    // logits tiles: A=Q[16x32c], B=K^T (columns = k rows of K storage)
    v16bf bk0 = load_tile_b(K, rowbase + k0, h * 32, 128, lane);
    v16bf bk1 = load_tile_b(K, rowbase + k0 + 16, h * 32, 128, lane);
    v8f S0 = {}, S1 = {};
    S0 = wmma_bf16(aq, bk0, S0);
    S1 = wmma_bf16(aq, bk1, S1);

    int kg0 = k0 + n, kg1 = k0 + 16 + n;
    float mv0 = ((transpose ? mask[(size_t)kg0 * NSEQ + i] : mask[(size_t)i * NSEQ + kg0]) - 1.0f) * 1e8f;
    float mv1 = ((transpose ? mask[(size_t)kg1 * NSEQ + i] : mask[(size_t)i * NSEQ + kg1]) - 1.0f) * 1e8f;

    float p0[8], p1[8];
#pragma unroll
    for (int j = 0; j < 8; ++j) {
      int qg = q0 + j + (hi << 3);
      float e0 = transpose ? eb[(size_t)h * NNPOS + (size_t)kg0 * NSEQ + qg]
                           : eb[(size_t)h * NNPOS + (size_t)qg * NSEQ + kg0];
      float e1 = transpose ? eb[(size_t)h * NNPOS + (size_t)kg1 * NSEQ + qg]
                           : eb[(size_t)h * NNPOS + (size_t)qg * NSEQ + kg1];
      float s0 = S0[j] * INV_SQRT_CH + e0 + mv0;
      float s1 = S1[j] * INV_SQRT_CH + e1 + mv1;
      float tm = hmax16(fmaxf(s0, s1));
      float nm = fmaxf(mrun[j], tm);
      float sc = __expf(mrun[j] - nm);
      mrun[j] = nm;
      float q0e = __expf(s0 - nm);
      float q1e = __expf(s1 - nm);
      lrun[j] = lrun[j] * sc + hsum16(q0e + q1e);
      O0[j] *= sc;
      O1[j] *= sc;
      p0[j] = q0e; p1[j] = q1e;
    }
    // stage P (bf16) through LDS, D-layout -> A-layout
#pragma unroll
    for (int j = 0; j < 8; ++j) {
      int m = j + (hi << 3);
      pbuf[slot][m][n] = (__bf16)p0[j];
      pbuf[slot][m][16 + n] = (__bf16)p1[j];
    }
    asm volatile("s_wait_dscnt 0" ::: "memory");
    v16bf ap = load_tile_a(&pbuf[slot][0][0], 0, 0, 32, lane);
    v16bf bv0 = load_tile_b(vbase, h * 32, k0, 256, lane);       // cols c=0..15
    v16bf bv1 = load_tile_b(vbase, h * 32 + 16, k0, 256, lane);  // cols c=16..31
    O0 = wmma_bf16(ap, bv0, O0);
    O1 = wmma_bf16(ap, bv1, O1);
  }

#pragma unroll
  for (int j = 0; j < 8; ++j) {
    int qg = q0 + j + (hi << 3);
    size_t r = (size_t)rowbase + qg;
    float inv = __builtin_amdgcn_rcpf(lrun[j]);
    int c0 = h * 32 + n, c1 = c0 + 16;
    float g0 = (float)G[r * 128 + c0];
    float g1 = (float)G[r * 128 + c1];
    OG[r * 128 + c0] = (__bf16)(O0[j] * inv * g0);
    OG[r * 128 + c1] = (__bf16)(O1[j] * inv * g1);
  }
}

extern "C" void kernel_launch(void* const* d_in, const int* in_sizes, int n_in,
                              void* d_out, int out_size, void* d_ws, size_t ws_size,
                              hipStream_t stream) {
  (void)in_sizes; (void)n_in; (void)out_size; (void)ws_size;
  const float* edge_embed = (const float*)d_in[0];
  const float* coords     = (const float*)d_in[1];
  const float* emask      = (const float*)d_in[2];
  const float* w1   = (const float*)d_in[3];
  const float* rg   = (const float*)d_in[4];
  const float* rb   = (const float*)d_in[5];
  const float* w2   = (const float*)d_in[6];
  const float* ln3g = (const float*)d_in[7];
  const float* ln3b = (const float*)d_in[8];
  const float* w3   = (const float*)d_in[9];

  char* ws = (char*)d_ws;
  const size_t NNc = 65536;
  size_t off = 0;
  float*  eb = (float*)(ws + off);   off += 4 * NNc * sizeof(float);
  __bf16* zn = (__bf16*)(ws + off);  off += NNc * 128 * 2;   // also reused as o*g output
  __bf16* qb = (__bf16*)(ws + off);  off += NNc * 128 * 2;
  __bf16* kb = (__bf16*)(ws + off);  off += NNc * 128 * 2;
  __bf16* vt = (__bf16*)(ws + off);  off += NNc * 128 * 2;
  __bf16* gb = (__bf16*)(ws + off);  off += NNc * 128 * 2;
  __bf16* wt = (__bf16*)(ws + off);  off += 5 * 128 * 128 * 2;
  float*  z1 = (float*)(ws + off);   off += NNc * 128 * sizeof(float);

  edge_bias_kernel<<<8192, 256, 0, stream>>>(coords, edge_embed, w1, rg, rb, w2, ln3g, ln3b, w3, eb);

  for (int pass = 0; pass < 2; ++pass) {
    int base = pass ? 17 : 10;
    const float* lng = (const float*)d_in[base + 0];
    const float* lnb = (const float*)d_in[base + 1];
    const float* wq  = (const float*)d_in[base + 2];
    const float* wk  = (const float*)d_in[base + 3];
    const float* wv  = (const float*)d_in[base + 4];
    const float* wg  = (const float*)d_in[base + 5];
    const float* wo  = (const float*)d_in[base + 6];

    prep_w<<<64, 256, 0, stream>>>(wq, wt + 0 * 16384);
    prep_w<<<64, 256, 0, stream>>>(wk, wt + 1 * 16384);
    prep_w<<<64, 256, 0, stream>>>(wv, wt + 2 * 16384);
    prep_w<<<64, 256, 0, stream>>>(wg, wt + 3 * 16384);
    prep_w<<<64, 256, 0, stream>>>(wo, wt + 4 * 16384);

    const float* zsrc = pass ? (const float*)z1 : edge_embed;
    ln_rows<<<8192, 256, 0, stream>>>(zsrc, lng, lnb, zn, pass);

    gemm_proj<<<1024, 256, 0, stream>>>(zn, wt + 0 * 16384, 0, qb, nullptr, nullptr, nullptr, 0);
    gemm_proj<<<1024, 256, 0, stream>>>(zn, wt + 1 * 16384, 0, kb, nullptr, nullptr, nullptr, 0);
    gemm_proj<<<1024, 256, 0, stream>>>(zn, wt + 2 * 16384, 1, nullptr, vt, nullptr, nullptr, 0);
    gemm_proj<<<1024, 256, 0, stream>>>(zn, wt + 3 * 16384, 2, gb, nullptr, nullptr, nullptr, 0);

    attn_kernel<<<2048, 256, 0, stream>>>(qb, kb, vt, gb, eb, emask, zn /*o*g reuses zn*/, pass);

    float* outp = pass ? (float*)d_out : z1;
    gemm_proj<<<1024, 256, 0, stream>>>(zn, wt + 4 * 16384, 3, nullptr, nullptr, zsrc, outp, pass);
  }
}
